// ScaledDotProductAttention_11630771438330
// MI455X (gfx1250) — matrix operand order repaired and checked by reading, compiled, not run on hardware
//
#include <hip/hip_runtime.h>

typedef __attribute__((ext_vector_type(16))) _Float16 v16h;
typedef __attribute__((ext_vector_type(8)))  _Float16 v8h;
typedef __attribute__((ext_vector_type(8)))  float    v8f;

#define S_LEN 2048
#define D_DIM 64
#define BC    64            // keys per inner iteration
#define NT    (S_LEN / BC)  // 32 key tiles
#define WAVES 8
#define BR    (WAVES*16)    // 128 queries per block
#define KSTR  72            // Ksh row stride in halves (row = 64 d + pad)
#define VSTR  72            // Vsh row stride in halves (row = 64 keys + pad)
#define PSTR  72            // Psh row stride in halves (row = 64 keys + pad)

// Butterfly max across each 16-lane group: single-instruction DPP16 steps
// (v_max_num_f32 with DPP on src0: dst = max(shuffle(x), x)).
__device__ __forceinline__ float red16_max(float x) {
    float y;
    asm("v_max_num_f32 %0, %1, %1 quad_perm:[1,0,3,2] row_mask:0xf bank_mask:0xf bound_ctrl:1"
        : "=v"(y) : "v"(x));
    asm("v_max_num_f32 %0, %1, %1 quad_perm:[2,3,0,1] row_mask:0xf bank_mask:0xf bound_ctrl:1"
        : "=v"(x) : "v"(y));
    asm("v_max_num_f32 %0, %1, %1 row_half_mirror row_mask:0xf bank_mask:0xf bound_ctrl:1"
        : "=v"(y) : "v"(x));
    asm("v_max_num_f32 %0, %1, %1 row_mirror row_mask:0xf bank_mask:0xf bound_ctrl:1"
        : "=v"(x) : "v"(y));
    return x;
}

__global__ __launch_bounds__(256) void fa_fwd_kernel(
    const float* __restrict__ Q,
    const float* __restrict__ K,
    const float* __restrict__ V,
    float* __restrict__ O)
{
    __shared__ _Float16 Ksh[2][D_DIM * KSTR];    // K tile: [key][d], double buffered
    __shared__ _Float16 Vsh[2][BC * VSTR];       // V tile, transposed: [d][key], double buffered
    __shared__ _Float16 Psh[WAVES][16 * PSTR];   // per-wave P transpose buffer

    const int tid  = threadIdx.x;
    const int lane = tid & 31;
    const int wave = tid >> 5;
    const int qtiles = S_LEN / BR;               // 16
    const int bh   = blockIdx.x / qtiles;
    const int qblk = blockIdx.x % qtiles;
    const int q0   = qblk * BR + wave * 16;

    const float* Qb = Q + (size_t)bh * S_LEN * D_DIM;
    const float* Kb = K + (size_t)bh * S_LEN * D_DIM;
    const float* Vb = V + (size_t)bh * S_LEN * D_DIM;
    float*       Ob = O + (size_t)bh * S_LEN * D_DIM;

    const int m16 = lane & 15;                   // M row (A/C layouts)
    const int g   = (lane & 16) ? 8 : 0;         // A-layout K-group offset

    // staging role: each thread owns (key = tid>>2, d = (tid&3)*16 .. +15)
    const int skey = tid >> 2;                   // 0..63
    const int sd   = (tid & 3) * 16;             // 0,16,32,48

    _Float16* const pw = &Psh[wave][0];

    // stage key-tile starting at key0 into LDS buffer `buf` (f32 -> f16, V transposed)
    auto stage = [&](int key0, int buf) {
        const float* ks = Kb + (size_t)(key0 + skey) * D_DIM + sd;
        #pragma unroll
        for (int q = 0; q < 4; ++q) {
            float4 a = ((const float4*)ks)[q];
            Ksh[buf][skey * KSTR + sd + 4*q + 0] = (_Float16)a.x;
            Ksh[buf][skey * KSTR + sd + 4*q + 1] = (_Float16)a.y;
            Ksh[buf][skey * KSTR + sd + 4*q + 2] = (_Float16)a.z;
            Ksh[buf][skey * KSTR + sd + 4*q + 3] = (_Float16)a.w;
        }
        const float* vs = Vb + (size_t)(key0 + skey) * D_DIM + sd;
        #pragma unroll
        for (int q = 0; q < 2; ++q) {
            float4 a = ((const float4*)vs)[2*q];
            float4 b = ((const float4*)vs)[2*q + 1];
            Vsh[buf][(sd + 8*q + 0) * VSTR + skey] = (_Float16)a.x;
            Vsh[buf][(sd + 8*q + 1) * VSTR + skey] = (_Float16)a.y;
            Vsh[buf][(sd + 8*q + 2) * VSTR + skey] = (_Float16)a.z;
            Vsh[buf][(sd + 8*q + 3) * VSTR + skey] = (_Float16)a.w;
            Vsh[buf][(sd + 8*q + 4) * VSTR + skey] = (_Float16)b.x;
            Vsh[buf][(sd + 8*q + 5) * VSTR + skey] = (_Float16)b.y;
            Vsh[buf][(sd + 8*q + 6) * VSTR + skey] = (_Float16)b.z;
            Vsh[buf][(sd + 8*q + 7) * VSTR + skey] = (_Float16)b.w;
        }
    };

    // ---- Load Q tile (16 x 64) as two f16 A-operands.
    // Fold 1/sqrt(64) AND log2(e) into Q so softmax runs in exp2 domain.
    v16h qa[2];
    {
        const float* qrow = Qb + (size_t)(q0 + m16) * D_DIM;
        const float scale = 0.125f * 1.44269504088896f;
        #pragma unroll
        for (int c = 0; c < 2; ++c) {
            #pragma unroll
            for (int e = 0; e < 8; ++e) {
                qa[c][e]     = (_Float16)(qrow[32*c + g      + e] * scale);
                qa[c][e + 8] = (_Float16)(qrow[32*c + 16 + g + e] * scale);
            }
        }
    }

    // all-ones B operand: softmax row-sums run on the matrix pipe
    v16h ones;
    #pragma unroll
    for (int e = 0; e < 16; ++e) ones[e] = (_Float16)1.0f;

    // ---- Accumulators & softmax state (base-2 domain)
    v8f o_acc[4];
    #pragma unroll
    for (int t = 0; t < 4; ++t)
        #pragma unroll
        for (int r = 0; r < 8; ++r) o_acc[t][r] = 0.0f;
    v8f l_acc;
    #pragma unroll
    for (int r = 0; r < 8; ++r) l_acc[r] = 0.0f;
    float m_r[8];
    #pragma unroll
    for (int r = 0; r < 8; ++r) m_r[r] = -INFINITY;

    // ---- Prologue: stage tile 0, prefetch tile 1
    stage(0, 0);
    {
        const float* kp = Kb + (size_t)(BC + skey) * D_DIM + sd;
        const float* vp = Vb + (size_t)(BC + skey) * D_DIM + sd;
        __builtin_prefetch(kp, 0, 3);
        __builtin_prefetch(vp, 0, 3);
    }
    __syncthreads();

    // ---- Main loop over key tiles of 64 (double buffered, one barrier/iter).
    // Unroll by 2 so `cur` is a compile-time constant and all LDS addresses
    // fold into immediate offsets (kills the per-iter v_mad/SALU address math).
    #pragma unroll 2
    for (int j = 0; j < NT; ++j) {
        const int cur = j & 1;

        // stage next tile into the other buffer; prefetch the one after
        if (j + 1 < NT) {
            stage((j + 1) * BC, cur ^ 1);
            if (j + 2 < NT) {
                const float* kp = Kb + (size_t)((j + 2) * BC + skey) * D_DIM + sd;
                const float* vp = Vb + (size_t)((j + 2) * BC + skey) * D_DIM + sd;
                __builtin_prefetch(kp, 0, 3);
                __builtin_prefetch(vp, 0, 3);
            }
        }

        // ---- B-operands for S = Q*K^T from [key][d] K rows in LDS
        v16h kb[4][2];
        #pragma unroll
        for (int c = 0; c < 2; ++c) {
            #pragma unroll
            for (int t = 0; t < 4; ++t) {
                const _Float16* krow = &Ksh[cur][(16*t + m16) * KSTR + 32*c];
                v8h lo = *(const v8h*)&krow[g];
                v8h hi = *(const v8h*)&krow[16 + g];
                #pragma unroll
                for (int e = 0; e < 8; ++e) { kb[t][c][e] = lo[e]; kb[t][c][e+8] = hi[e]; }
            }
        }

        // ---- Scores: four 16x16 f32 tiles
        v8f s[4];
        #pragma unroll
        for (int t = 0; t < 4; ++t) {
            v8f acc = {};
            acc = __builtin_amdgcn_wmma_f32_16x16x32_f16(false, qa[0], false, kb[t][0], (short)0, acc, false, false);
            acc = __builtin_amdgcn_wmma_f32_16x16x32_f16(false, qa[1], false, kb[t][1], (short)0, acc, false, false);
            s[t] = acc;
        }

        // ---- Online softmax (base-2 domain); max via DPP16, sum via WMMA below
        float pr[4][8];
        #pragma unroll
        for (int r = 0; r < 8; ++r) {
            float smax = fmaxf(fmaxf(s[0][r], s[1][r]), fmaxf(s[2][r], s[3][r]));
            smax = red16_max(smax);
            float mnew  = fmaxf(m_r[r], smax);
            float alpha = exp2f(m_r[r] - mnew);
            m_r[r] = mnew;
            pr[0][r] = exp2f(s[0][r] - mnew);
            pr[1][r] = exp2f(s[1][r] - mnew);
            pr[2][r] = exp2f(s[2][r] - mnew);
            pr[3][r] = exp2f(s[3][r] - mnew);
            l_acc[r]    *= alpha;
            o_acc[0][r] *= alpha; o_acc[1][r] *= alpha;
            o_acc[2][r] *= alpha; o_acc[3][r] *= alpha;
        }

        // ---- Transpose P through LDS: C layout (lane=key) -> A layout (lane=row)
        #pragma unroll
        for (int r = 0; r < 8; ++r) {
            const int mrow = r + ((lane & 16) ? 8 : 0);
            #pragma unroll
            for (int t = 0; t < 4; ++t)
                pw[mrow * PSTR + 16*t + m16] = (_Float16)pr[t][r];
        }
        asm volatile("s_wait_dscnt 0" ::: "memory");

        // Two A-operands of P: keys 0-31 and keys 32-63
        v16h pa[2];
        #pragma unroll
        for (int cc = 0; cc < 2; ++cc) {
            v8h lo = *(const v8h*)&pw[m16 * PSTR + 32*cc + g];
            v8h hi = *(const v8h*)&pw[m16 * PSTR + 32*cc + 16 + g];
            #pragma unroll
            for (int e = 0; e < 8; ++e) { pa[cc][e] = lo[e]; pa[cc][e + 8] = hi[e]; }
        }

        // ---- l += rowsum(P): P x ones on the matrix pipe (broadcast to all cols)
        l_acc = __builtin_amdgcn_wmma_f32_16x16x32_f16(false, pa[0], false, ones, (short)0, l_acc, false, false);
        l_acc = __builtin_amdgcn_wmma_f32_16x16x32_f16(false, pa[1], false, ones, (short)0, l_acc, false, false);

        // ---- O += P * V  (V operands built after scores so kb registers are dead)
        #pragma unroll
        for (int cc = 0; cc < 2; ++cc) {
            #pragma unroll
            for (int t2 = 0; t2 < 4; ++t2) {
                const _Float16* vrow = &Vsh[cur][(16*t2 + m16) * VSTR + 32*cc];
                v16h vb;
                v8h lo = *(const v8h*)&vrow[g];
                v8h hi = *(const v8h*)&vrow[16 + g];
                #pragma unroll
                for (int e = 0; e < 8; ++e) { vb[e] = lo[e]; vb[e+8] = hi[e]; }
                o_acc[t2] = __builtin_amdgcn_wmma_f32_16x16x32_f16(false, pa[cc], false, vb, (short)0, o_acc[t2], false, false);
            }
        }

        __syncthreads();   // next tile staged + this tile consumed
    }

    // ---- Normalize and store (C layout)
    #pragma unroll
    for (int r = 0; r < 8; ++r) {
        const float inv  = 1.0f / l_acc[r];
        const int   mrow = r + ((lane & 16) ? 8 : 0);
        float* orow = Ob + (size_t)(q0 + mrow) * D_DIM;
        #pragma unroll
        for (int t2 = 0; t2 < 4; ++t2)
            orow[16 * t2 + m16] = o_acc[t2][r] * inv;
    }
}

extern "C" void kernel_launch(void* const* d_in, const int* in_sizes, int n_in,
                              void* d_out, int out_size, void* d_ws, size_t ws_size,
                              hipStream_t stream) {
    const float* Q = (const float*)d_in[0];
    const float* K = (const float*)d_in[1];
    const float* V = (const float*)d_in[2];
    float*       O = (float*)d_out;

    const int bh = in_sizes[0] / (S_LEN * D_DIM);   // batch * heads = 32
    dim3 grid(bh * (S_LEN / BR));                   // 512 blocks
    dim3 block(256);                                // 8 wave32 waves
    fa_fwd_kernel<<<grid, block, 0, stream>>>(Q, K, V, O);
}
